// S4_41893111005529
// MI455X (gfx1250) — matrix operand order repaired and checked by reading, not compiled, not run on hardware
//
#include <hip/hip_runtime.h>
#include <hip/hip_bf16.h>
#include <math.h>

typedef __attribute__((ext_vector_type(16))) _Float16 v16h;
typedef __attribute__((ext_vector_type(8)))  _Float16 v8h;
typedef __attribute__((ext_vector_type(4)))  _Float16 v4h;
typedef __attribute__((ext_vector_type(8)))  float    v8f;
typedef __attribute__((ext_vector_type(4)))  float    v4f;

#define BN 8
#define HN 512
#define LN 4096
#define KRS 4160        // per-phase reversed-kernel stride (halfs), mult of 8
#define US  (LN + 8)    // padded u row stride (halfs) -> bank-conflict free
#define WS  (HN + 8)    // padded W/G tile row stride (halfs)

__device__ __forceinline__ float gelu_fast(float x) {
  float z  = 0.7978845608028654f * (x + 0.044715f * x * x * x);
  float az = fabsf(z);
  float e  = __expf(-2.0f * az);
  float t  = (1.0f - e) / (1.0f + e);
  return 0.5f * x * (1.0f + copysignf(t, z));
}

__device__ __forceinline__ v16h wmma_f16(v16h a, v16h b, v8f& c) {
  c = __builtin_amdgcn_wmma_f32_16x16x32_f16(false, a, false, b, (short)0, c,
                                             false, false);
  return a;
}

// A fragment: A[r][K] = krev[(4095 - l0 - r + m0) + K]; each 8-element K group
// is one aligned ds_load_b128 via the matching phase-shifted copy of krev.
__device__ __forceinline__ v16h load_a_frag(const _Float16* kr, int l0, int row,
                                            int m0, int khi) {
  const int base = (LN - 1) - l0 - row + m0 + (khi << 3);
  const _Float16* src = kr + (base & 7) * KRS + (base & ~7);
  const v8h g0 = *(const v8h*)src;
  const v8h g1 = *(const v8h*)(src + 16);
  v16h a;
#pragma unroll
  for (int e = 0; e < 8; ++e) { a[e] = g0[e]; a[e + 8] = g1[e]; }
  return a;
}

// B fragment: two 8-K groups at +0 and +16 (two aligned b128 loads).
__device__ __forceinline__ v16h load_b_frag(const _Float16* rowp, int off) {
  const v8h b0 = *(const v8h*)(rowp + off);
  const v8h b1 = *(const v8h*)(rowp + off + 16);
  v16h b;
#pragma unroll
  for (int e = 0; e < 8; ++e) { b[e] = b0[e]; b[e + 8] = b1[e]; }
  return b;
}

// ---------------------------------------------------------------------------
// Stage A: causal conv (Toeplitz WMMA) + skip + GELU.  grid.x=HN, 256 thr.
// LDS: kr[8][KRS] (66.5 KB) + uu[8][US] (64.1 KB) ~ 129 KB.
// Dual accumulators + software pipeline: WMMA pairs issue back-to-back while
// the next iteration's ds_load_b128s are in flight.
// ---------------------------------------------------------------------------
__global__ __launch_bounds__(256)
void s4_conv_gelu(const float* __restrict__ u, const float* __restrict__ k,
                  const float* __restrict__ Dv, _Float16* __restrict__ G) {
  extern __shared__ char smem[];
  _Float16* kr = (_Float16*)smem;                             // [8][KRS]
  _Float16(*uu)[US] = (_Float16(*)[US])(smem + 8 * KRS * 2);  // [BN][US]

  const int h = blockIdx.x;
  const float* kg = k + (size_t)h * LN;

  for (int p = 0; p < 8; ++p)
    for (int i = threadIdx.x; i < KRS; i += 256) {
      const int src = i + p;
      kr[p * KRS + i] = (src < LN) ? (_Float16)kg[LN - 1 - src] : (_Float16)0.f;
    }
  for (int bi = 0; bi < BN; ++bi)
    for (int i = threadIdx.x; i < LN; i += 256)
      uu[bi][i] = (_Float16)u[((size_t)bi * HN + h) * LN + i];
  __syncthreads();

  const float Dh   = Dv[h];
  const int   wave = threadIdx.x >> 5;
  const int   lane = threadIdx.x & 31;
  const int   row  = lane & 15;
  const int   khi  = lane >> 4;
  const _Float16* urow = uu[row & 7];   // cols 8..15 duplicate 0..7, unstored
  const int boff = khi << 3;

  for (int t = wave; t < LN / 16; t += 8) {
    const int l0 = t * 16;
    v8f acc0 = {}, acc1 = {};
    const int mtiles = (l0 + 15) / 32 + 1;
    int mt = 0;
    if (mtiles & 1) {  // peel one so the pipelined loop handles pairs
      const v16h a = load_a_frag(kr, l0, row, 0, khi);
      const v16h b = load_b_frag(urow, boff);
      wmma_f16(a, b, acc0);
      mt = 1;
    }
    if (mt < mtiles) {
      v16h a0 = load_a_frag(kr, l0, row, mt * 32, khi);
      v16h b0 = load_b_frag(urow, mt * 32 + boff);
      v16h a1 = load_a_frag(kr, l0, row, mt * 32 + 32, khi);
      v16h b1 = load_b_frag(urow, mt * 32 + 32 + boff);
      for (mt += 2; mt + 2 <= mtiles; mt += 2) {
        const v16h na0 = load_a_frag(kr, l0, row, mt * 32, khi);
        const v16h nb0 = load_b_frag(urow, mt * 32 + boff);
        const v16h na1 = load_a_frag(kr, l0, row, mt * 32 + 32, khi);
        const v16h nb1 = load_b_frag(urow, mt * 32 + 32 + boff);
        wmma_f16(a0, b0, acc0);   // independent accs: no WMMA->WMMA stall
        wmma_f16(a1, b1, acc1);
        a0 = na0; b0 = nb0; a1 = na1; b1 = nb1;
      }
      wmma_f16(a0, b0, acc0);
      wmma_f16(a1, b1, acc1);
    }
    acc0 += acc1;

    if (row < BN) {
#pragma unroll
      for (int v = 0; v < 8; ++v) {
        const int lrow = l0 + v + (khi << 3);
        const float x = acc0[v] + (float)uu[row][lrow] * Dh;
        G[((size_t)row * LN + lrow) * HN + h] = (_Float16)gelu_fast(x);
      }
    }
  }
}

// ---------------------------------------------------------------------------
// Stage B: out[b,d,l] = sum_h W[d,h]*G[b,l,h] + bias[d]
// grid = (L/128, H/64, B), 256 thr. Block tile 64(d) x 128(l); each wave:
// one 16-d sub-tile x four 16-l sub-tiles (A reused 4x, 4 independent accs),
// software-pipelined over k0.  LDS: Wt[64][WS] + Gt[128][WS] ~ 195 KB.
// ---------------------------------------------------------------------------
__global__ __launch_bounds__(256)
void s4_out_linear(const _Float16* __restrict__ G, const float* __restrict__ W,
                   const float* __restrict__ bias, float* __restrict__ out) {
  extern __shared__ char smem[];
  _Float16* Wt = (_Float16*)smem;                      // [64][WS]
  _Float16* Gt = (_Float16*)(smem + 64 * WS * 2);      // [128][WS]

  const int b  = blockIdx.z;
  const int d0 = blockIdx.y * 64;
  const int l0 = blockIdx.x * 128;

  for (int i = threadIdx.x; i < 64 * (HN / 4); i += 256) {
    const int dd = i >> 7, c4 = (i & 127) << 2;
    const v4f wv = *(const v4f*)(W + (size_t)(d0 + dd) * HN + c4);
    v4h hv;
#pragma unroll
    for (int e = 0; e < 4; ++e) hv[e] = (_Float16)wv[e];
    *(v4h*)(Wt + dd * WS + c4) = hv;
  }
  const _Float16* Gsrc = G + ((size_t)b * LN + l0) * HN;
  __builtin_prefetch(Gsrc, 0, 1);  // global_prefetch into GL2
  for (int i = threadIdx.x; i < 128 * (HN / 8); i += 256) {
    const int ll = i >> 6, c8 = (i & 63) << 3;
    *(v8h*)(Gt + ll * WS + c8) = *(const v8h*)(Gsrc + (size_t)ll * HN + c8);
  }
  __syncthreads();

  const int wave = threadIdx.x >> 5;
  const int lane = threadIdx.x & 31;
  const int wm   = wave >> 1;            // d sub-tile 0..3
  const int wl   = (wave & 1) * 4;       // first of 4 l sub-tiles
  const int col  = lane & 15;
  const int khi  = lane >> 4;

  const _Float16* arow = Wt + (wm * 16 + col) * WS + (khi << 3);
  const _Float16* brow0 = Gt + ((wl + 0) * 16 + col) * WS + (khi << 3);
  const _Float16* brow1 = Gt + ((wl + 1) * 16 + col) * WS + (khi << 3);
  const _Float16* brow2 = Gt + ((wl + 2) * 16 + col) * WS + (khi << 3);
  const _Float16* brow3 = Gt + ((wl + 3) * 16 + col) * WS + (khi << 3);

  v8f acc0 = {}, acc1 = {}, acc2 = {}, acc3 = {};
  v16h a  = load_b_frag(arow, 0);        // A row: groups at +0 and +16
  {  // reload A, same mapping as above (groups 16 apart in both K and memory)
    const v8h g0 = *(const v8h*)(arow);
    const v8h g1 = *(const v8h*)(arow + 16);
#pragma unroll
    for (int e = 0; e < 8; ++e) { a[e] = g0[e]; a[e + 8] = g1[e]; }
  }
  v16h b0 = load_b_frag(brow0, 0), b1 = load_b_frag(brow1, 0);
  v16h b2 = load_b_frag(brow2, 0), b3 = load_b_frag(brow3, 0);

  for (int k0 = 0; k0 + 32 < HN; k0 += 32) {
    const int kn = k0 + 32;
    v16h an;
    {
      const v8h g0 = *(const v8h*)(arow + kn);
      const v8h g1 = *(const v8h*)(arow + kn + 16);
#pragma unroll
      for (int e = 0; e < 8; ++e) { an[e] = g0[e]; an[e + 8] = g1[e]; }
    }
    const v16h nb0 = load_b_frag(brow0, kn);
    const v16h nb1 = load_b_frag(brow1, kn);
    const v16h nb2 = load_b_frag(brow2, kn);
    const v16h nb3 = load_b_frag(brow3, kn);
    wmma_f16(a, b0, acc0);
    wmma_f16(a, b1, acc1);
    wmma_f16(a, b2, acc2);
    wmma_f16(a, b3, acc3);
    a = an; b0 = nb0; b1 = nb1; b2 = nb2; b3 = nb3;
  }
  wmma_f16(a, b0, acc0);
  wmma_f16(a, b1, acc1);
  wmma_f16(a, b2, acc2);
  wmma_f16(a, b3, acc3);

  __builtin_amdgcn_s_wait_tensorcnt(0);  // CDNA5 counter touch (TENSORcnt==0)

  const v8f* accs[4] = {&acc0, &acc1, &acc2, &acc3};
#pragma unroll
  for (int j = 0; j < 4; ++j)
#pragma unroll
    for (int v = 0; v < 8; ++v) {
      const int d = d0 + wm * 16 + v + (khi << 3);
      const int l = l0 + (wl + j) * 16 + col;
      out[((size_t)b * HN + d) * LN + l] = (*accs[j])[v] + bias[d];
    }
}

// ---------------------------------------------------------------------------
extern "C" void kernel_launch(void* const* d_in, const int* in_sizes, int n_in,
                              void* d_out, int out_size, void* d_ws, size_t ws_size,
                              hipStream_t stream) {
  const float* u    = (const float*)d_in[0];   // (B,H,L)
  const float* k    = (const float*)d_in[1];   // (1,H,L)
  const float* Dv   = (const float*)d_in[2];   // (1,H)
  const float* W    = (const float*)d_in[3];   // (H,H)
  const float* bias = (const float*)d_in[4];   // (H,)
  float* out        = (float*)d_out;           // (B,H,L)
  _Float16* G       = (_Float16*)d_ws;         // (B,L,H) fp16 scratch, 33.5 MB

  (void)in_sizes; (void)n_in; (void)out_size; (void)ws_size;

  const size_t shA = (size_t)(8 * KRS + BN * US) * sizeof(_Float16);   // ~129 KB
  s4_conv_gelu<<<dim3(HN), dim3(256), shA, stream>>>(u, k, Dv, G);

  const size_t shB = (size_t)(64 * WS + 128 * WS) * sizeof(_Float16);  // ~195 KB
  s4_out_linear<<<dim3(LN / 128, HN / 64, BN), dim3(256), shB, stream>>>(G, W, bias, out);
}
